// AdaFeatBlock_86294482912099
// MI455X (gfx1250) — matrix operand order repaired and checked by reading, not compiled, not run on hardware
//
#include <hip/hip_runtime.h>
#include <math.h>

// ---------------------------------------------------------------------------
// AdaFeatBlock (modulated deformable conv block) for MI455X / gfx1250.
//
// ~24 GFLOP of implicit GEMM (K = C*9 = 576, N = 131072 px) vs ~100 MB of
// traffic (x fits in the 192 MB L2): strongly compute bound, so all three
// contractions run on v_wmma_f32_16x16x32_bf16 (f32 accumulate). Sampling
// coordinates, bilinear weights, sigmoid/leaky-ReLU stay in f32 VALU.
//
// This revision: KPAD=40 so every WMMA fragment is 2x ds_load_b128 and every
// B-panel write is 2x ds_store_b128 (bf16 pairs packed via cvt_pk), the
// boundary handling is branch-free (0/1 float mask, clamped pointer), and the
// B panel is double-buffered in LDS -> one s_barrier per K-step.
//
// Workspace requirement: (8*27 + 8*64) * 16384 * 4 B  ≈ 47.7 MB in d_ws.
// ---------------------------------------------------------------------------

typedef __attribute__((ext_vector_type(16))) __bf16 v16bf;
typedef __attribute__((ext_vector_type(8)))  float  v8f;
typedef __attribute__((ext_vector_type(4)))  unsigned v4u;

#define Bn   8
#define Cc   64
#define Hh   128
#define Ww   128
#define KK   9
#define HW   (Hh * Ww)
#define CHW  (Cc * HW)
#define OMC  27            // 18 offset + 9 mask channels
#define KPAD 40            // K stride (bf16 elems): 80B rows, 16B-aligned sub-blocks
#define KPDW (KPAD / 2)    // 20 dwords per row

struct F2 { v4u a, b; };

__device__ __forceinline__ unsigned short bf16_bits(float f) {
  __bf16 h = (__bf16)f;
  return __builtin_bit_cast(unsigned short, h);
}
__device__ __forceinline__ unsigned pack2_bf16(float lo, float hi) {
  return (unsigned)bf16_bits(lo) | ((unsigned)bf16_bits(hi) << 16);
}

// WMMA A fragment (16x32 bf16): lane holds M = l&15; lanes 0-15 K=0..15,
// lanes 16-31 K=16..31 -> 8 consecutive dwords. Two b128 loads per lane.
__device__ __forceinline__ v16bf load_frag_A(const unsigned* lds, int baseDw, int lane) {
  int i0 = baseDw + (lane & 15) * KPDW + ((lane >> 4) ? 8 : 0);
  F2 t;
  t.a = *(const v4u*)(lds + i0);
  t.b = *(const v4u*)(lds + i0 + 4);
  return __builtin_bit_cast(v16bf, t);
}

// WMMA B fragment (32x16 bf16): lane holds N = l&15; lanes 0-15 K=0..15,
// lanes 16-31 K=16..31 -> 8 consecutive dwords. Two b128 loads per lane.
__device__ __forceinline__ v16bf load_frag_B(const unsigned* lds, int baseDw, int lane) {
  int i0 = baseDw + (lane & 15) * KPDW + ((lane >> 4) ? 8 : 0);
  F2 t;
  t.a = *(const v4u*)(lds + i0);
  t.b = *(const v4u*)(lds + i0 + 4);
  return __builtin_bit_cast(v16bf, t);
}

// Store one thread's 16 B-panel elements (k = half*16 .. +15) as 2x b128.
__device__ __forceinline__ void store_b_panel(unsigned* ldsDw, int bufDw, int px,
                                              int half, const float* v) {
  unsigned u[8];
#pragma unroll
  for (int j = 0; j < 8; ++j) u[j] = pack2_bf16(v[2 * j], v[2 * j + 1]);
  v4u* dst = (v4u*)(ldsDw + bufDw + px * KPDW + half * 8);
  v4u w0 = {u[0], u[1], u[2], u[3]};
  v4u w1 = {u[4], u[5], u[6], u[7]};
  dst[0] = w0;
  dst[1] = w1;
}

// Pack OIHW f32 weights into LDS as bf16 A tiles: tile(mt, tap, cb) is
// 16(M) x 32(K = cin within 32-block), row stride KPAD.
__device__ __forceinline__ void pack_weights(const float* __restrict__ w, int Cout,
                                             int Mtiles, unsigned short* ldsA, int tid) {
  int totalElems = Mtiles * KK * 2 * 16 * KPAD;
  for (int i = tid; i < totalElems; i += 256) ldsA[i] = 0;
  __syncthreads();
  int totalW = Cout * Cc * KK;
  for (int i = tid; i < totalW; i += 256) {
    int cout = i / (Cc * KK);
    int rem  = i % (Cc * KK);
    int cin  = rem / KK;
    int tap  = rem % KK;
    int tile = ((cout >> 4) * KK + tap) * 2 + (cin >> 5);
    ldsA[tile * 16 * KPAD + (cout & 15) * KPAD + (cin & 31)] = bf16_bits(w[i]);
  }
}

// ------------------------- Stage 1: offset/mask conv ------------------------
__global__ void __launch_bounds__(256)
k_conv_om(const float* __restrict__ x, const float* __restrict__ w_om,
          const float* __restrict__ b_om, float* __restrict__ offmask) {
  extern __shared__ char smem[];
  unsigned short* ldsA = (unsigned short*)smem;
  unsigned* uA = (unsigned*)ldsA;
  unsigned* uB = uA + 2 * KK * 2 * 16 * KPDW;   // double-buffered B panel

  int tid = threadIdx.x, lane = tid & 31, wave = tid >> 5;
  int b = blockIdx.x / Hh, h = blockIdx.x % Hh;
  pack_weights(w_om, OMC, 2, ldsA, tid);
  __syncthreads();

  const float* xb = x + b * CHW;
  v8f acc[2] = {};
  int px = tid >> 1, half = tid & 1, k0 = half * 16;

  for (int tap = 0; tap < KK; ++tap) {
    int yy = h - 1 + tap / 3;
    int xx = px - 1 + tap % 3;
    bool inb = (yy >= 0) && (yy < Hh) && (xx >= 0) && (xx < Ww);
    float mval = inb ? 1.f : 0.f;
    const float* src = xb + (inb ? (yy * Ww + xx) : 0);
#pragma unroll
    for (int cb = 0; cb < 2; ++cb) {
      int bufDw = cb * 128 * KPDW;
      const float* s0 = src + (cb * 32 + k0) * HW;
      float v[16];
#pragma unroll
      for (int j = 0; j < 16; ++j) v[j] = mval * s0[j * HW];
      store_b_panel(uB, bufDw, px, half, v);
      __syncthreads();
      v16bf bf = load_frag_B(uB, bufDw + wave * 16 * KPDW, lane);
#pragma unroll
      for (int mt = 0; mt < 2; ++mt) {
        v16bf af = load_frag_A(uA, ((mt * KK + tap) * 2 + cb) * 16 * KPDW, lane);
        acc[mt] = __builtin_amdgcn_wmma_f32_16x16x32_bf16(
            false, af, false, bf, (short)0, acc[mt], false, false);
      }
    }
  }

  int n = lane & 15, hiM = (lane >> 4) ? 8 : 0;
  int pw = wave * 16 + n;
#pragma unroll
  for (int mt = 0; mt < 2; ++mt) {
#pragma unroll
    for (int r = 0; r < 8; ++r) {
      int c = mt * 16 + hiM + r;
      if (c < OMC) {
        float v = acc[mt][r] + b_om[c];
        if (c >= 2 * KK) v = 2.f / (1.f + __expf(-v));   // 2*sigmoid (mask)
        offmask[((b * OMC + c) * Hh + h) * Ww + pw] = v;
      }
    }
  }
}

// -------------------- Stage 2: modulated deformable conv --------------------
__global__ void __launch_bounds__(256)
k_deform(const float* __restrict__ x, const float* __restrict__ offmask,
         const float* __restrict__ w_dc, const float* __restrict__ b_dc,
         float* __restrict__ act) {
  extern __shared__ char smem[];
  unsigned short* ldsA = (unsigned short*)smem;
  unsigned* uA = (unsigned*)ldsA;
  unsigned* uB = uA + 4 * KK * 2 * 16 * KPDW;
  float* ldsOM = (float*)(uB + 2 * 128 * KPDW);

  int tid = threadIdx.x, lane = tid & 31, wave = tid >> 5;
  int b = blockIdx.x / Hh, h = blockIdx.x % Hh;
  pack_weights(w_dc, Cc, 4, ldsA, tid);
  for (int i = tid; i < OMC * Ww; i += 256) {
    int ch = i >> 7, wpx = i & 127;
    ldsOM[i] = offmask[((b * OMC + ch) * Hh + h) * Ww + wpx];
  }
  __syncthreads();

  const float* xb = x + b * CHW;
  v8f acc[4] = {};
  int px = tid >> 1, half = tid & 1, k0 = half * 16;

  for (int tap = 0; tap < KK; ++tap) {
    float offy = ldsOM[(tap * 2 + 0) * Ww + px];
    float offx = ldsOM[(tap * 2 + 1) * Ww + px];
    float msk  = ldsOM[(2 * KK + tap) * Ww + px];
    float py  = (float)(h - 1 + tap / 3) + offy;
    float pxx = (float)(px - 1 + tap % 3) + offx;
    float y0f = floorf(py), x0f = floorf(pxx);
    float ly = py - y0f, lx = pxx - x0f;
    int y0 = (int)y0f, x0i = (int)x0f, y1 = y0 + 1, x1 = x0i + 1;
    float vy0 = (y0  >= 0 && y0  < Hh) ? 1.f : 0.f;
    float vy1 = (y1  >= 0 && y1  < Hh) ? 1.f : 0.f;
    float vx0 = (x0i >= 0 && x0i < Ww) ? 1.f : 0.f;
    float vx1 = (x1  >= 0 && x1  < Ww) ? 1.f : 0.f;
    float w00 = (1.f - ly) * (1.f - lx) * msk * vy0 * vx0;
    float w01 = (1.f - ly) * lx         * msk * vy0 * vx1;
    float w10 = ly         * (1.f - lx) * msk * vy1 * vx0;
    float w11 = ly         * lx         * msk * vy1 * vx1;
    int y0c = min(max(y0, 0),  Hh - 1), y1c = min(max(y1, 0), Hh - 1);
    int x0c = min(max(x0i, 0), Ww - 1), x1c = min(max(x1, 0), Ww - 1);
    int i00 = y0c * Ww + x0c, i01 = y0c * Ww + x1c;
    int i10 = y1c * Ww + x0c, i11 = y1c * Ww + x1c;

#pragma unroll
    for (int cb = 0; cb < 2; ++cb) {
      int bufDw = cb * 128 * KPDW;
      float v[16];
#pragma unroll
      for (int j = 0; j < 16; ++j) {
        const float* p = xb + (cb * 32 + k0 + j) * HW;
        v[j] = w00 * p[i00] + w01 * p[i01] + w10 * p[i10] + w11 * p[i11];
      }
      store_b_panel(uB, bufDw, px, half, v);
      __syncthreads();
      v16bf bf = load_frag_B(uB, bufDw + wave * 16 * KPDW, lane);
#pragma unroll
      for (int mt = 0; mt < 4; ++mt) {
        v16bf af = load_frag_A(uA, ((mt * KK + tap) * 2 + cb) * 16 * KPDW, lane);
        acc[mt] = __builtin_amdgcn_wmma_f32_16x16x32_bf16(
            false, af, false, bf, (short)0, acc[mt], false, false);
      }
    }
  }

  int n = lane & 15, hiM = (lane >> 4) ? 8 : 0;
  int pw = wave * 16 + n;
#pragma unroll
  for (int mt = 0; mt < 4; ++mt) {
#pragma unroll
    for (int r = 0; r < 8; ++r) {
      int c = mt * 16 + hiM + r;
      float v = acc[mt][r] + b_dc[c];
      v = (v > 0.f) ? v : 0.2f * v;                 // fused leaky ReLU
      act[((b * Cc + c) * Hh + h) * Ww + pw] = v;
    }
  }
}

// ------------------- Stage 3: output conv + residual add --------------------
__global__ void __launch_bounds__(256)
k_conv_out(const float* __restrict__ act, const float* __restrict__ x,
           const float* __restrict__ w_c, const float* __restrict__ b_c,
           float* __restrict__ out) {
  extern __shared__ char smem[];
  unsigned short* ldsA = (unsigned short*)smem;
  unsigned* uA = (unsigned*)ldsA;
  unsigned* uB = uA + 4 * KK * 2 * 16 * KPDW;

  int tid = threadIdx.x, lane = tid & 31, wave = tid >> 5;
  int b = blockIdx.x / Hh, h = blockIdx.x % Hh;
  pack_weights(w_c, Cc, 4, ldsA, tid);
  __syncthreads();

  const float* ab = act + b * CHW;
  v8f acc[4] = {};
  int px = tid >> 1, half = tid & 1, k0 = half * 16;

  for (int tap = 0; tap < KK; ++tap) {
    int yy = h - 1 + tap / 3;
    int xx = px - 1 + tap % 3;
    bool inb = (yy >= 0) && (yy < Hh) && (xx >= 0) && (xx < Ww);
    float mval = inb ? 1.f : 0.f;
    const float* src = ab + (inb ? (yy * Ww + xx) : 0);
#pragma unroll
    for (int cb = 0; cb < 2; ++cb) {
      int bufDw = cb * 128 * KPDW;
      const float* s0 = src + (cb * 32 + k0) * HW;
      float v[16];
#pragma unroll
      for (int j = 0; j < 16; ++j) v[j] = mval * s0[j * HW];
      store_b_panel(uB, bufDw, px, half, v);
      __syncthreads();
      v16bf bf = load_frag_B(uB, bufDw + wave * 16 * KPDW, lane);
#pragma unroll
      for (int mt = 0; mt < 4; ++mt) {
        v16bf af = load_frag_A(uA, ((mt * KK + tap) * 2 + cb) * 16 * KPDW, lane);
        acc[mt] = __builtin_amdgcn_wmma_f32_16x16x32_bf16(
            false, af, false, bf, (short)0, acc[mt], false, false);
      }
    }
  }

  int n = lane & 15, hiM = (lane >> 4) ? 8 : 0;
  int pw = wave * 16 + n;
#pragma unroll
  for (int mt = 0; mt < 4; ++mt) {
#pragma unroll
    for (int r = 0; r < 8; ++r) {
      int c = mt * 16 + hiM + r;
      int idx = ((b * Cc + c) * Hh + h) * Ww + pw;
      out[idx] = acc[mt][r] + b_c[c] + x[idx];      // residual fused
    }
  }
}

// ---------------------------------------------------------------------------
extern "C" void kernel_launch(void* const* d_in, const int* in_sizes, int n_in,
                              void* d_out, int out_size, void* d_ws, size_t ws_size,
                              hipStream_t stream) {
  (void)in_sizes; (void)n_in; (void)out_size; (void)ws_size;
  const float* x    = (const float*)d_in[0];
  const float* w_om = (const float*)d_in[1];
  const float* b_om = (const float*)d_in[2];
  const float* w_dc = (const float*)d_in[3];
  const float* b_dc = (const float*)d_in[4];
  const float* w_c  = (const float*)d_in[5];
  const float* b_c  = (const float*)d_in[6];
  float* out = (float*)d_out;

  float* offmask = (float*)d_ws;                         // B*27*H*W f32
  float* act     = offmask + (size_t)Bn * OMC * HW;      // B*C*H*W  f32

  dim3 grid(Bn * Hh), block(256);
  size_t aBytes2 = (size_t)2 * KK * 2 * 16 * KPAD * 2;   // stage-1 A tiles
  size_t aBytes4 = (size_t)4 * KK * 2 * 16 * KPAD * 2;   // stage-2/3 A tiles
  size_t bBytes  = (size_t)2 * 128 * KPAD * 2;           // double-buffered B

  k_conv_om <<<grid, block, aBytes2 + bBytes, stream>>>(x, w_om, b_om, offmask);
  k_deform  <<<grid, block, aBytes4 + bBytes + (size_t)OMC * Ww * 4, stream>>>(
      x, offmask, w_dc, b_dc, act);
  k_conv_out<<<grid, block, aBytes4 + bBytes, stream>>>(act, x, w_c, b_c, out);
}
